// DeChunking_13709535609071
// MI455X (gfx1250) — matrix operand order repaired and checked by reading, not compiled, not run on hardware
//
#include <hip/hip_runtime.h>
#include <hip/hip_bf16.h>

typedef __attribute__((ext_vector_type(16))) _Float16 v16h;
typedef __attribute__((ext_vector_type(8)))  float    v8f;

#define EPS_ 1e-12f
#define B_   2
#define L_   4096
#define D_   768
#define T_   32            // chunk length (= WMMA K)
#define NC_  (L_ / T_)     // 128 chunks per sequence

// ---------------------------------------------------------------------------
// Kernel 1: per-chunk local EMA via WMMA.
//   y_local(32 x D) = W_chunk(32x32, lower-tri) @ z_chunk(32 x D)
// Done as two v_wmma_f32_16x16x32_f16 per 16-wide D tile (top/bottom rows).
// Also emits: Pin[b,l]  = prod_{k=chunk_start..l} decay_k   (f32)
//             Ptot[b,c] = chunk total decay product          (f32)
//             yend[b,c,d] = y_local at last row of chunk     (f32)
// ---------------------------------------------------------------------------
__global__ __launch_bounds__(256)
void ema_chunk_kernel(const float* __restrict__ z, const float* __restrict__ pt,
                      float* __restrict__ out,
                      float* __restrict__ Pin, float* __restrict__ Ptot,
                      float* __restrict__ yend)
{
    __shared__ float s_S[T_];   // inclusive cumsum of log(decay) within chunk
    __shared__ float s_p[T_];   // p_j
    __shared__ float s_P[T_];   // inclusive decay product within chunk

    const int bc = blockIdx.x;          // b * NC + c
    const int b  = bc / NC_;
    const int c  = bc % NC_;
    const int l0 = c * T_;

    const int tid  = threadIdx.x;
    const int lane = tid & 31;
    const int wave = tid >> 5;          // 0..7
    const int hi   = lane >> 4;         // lane half
    const int lo   = lane & 15;

    // --- chunk scalars (first 32 threads), tiny serial cumsum by thread 0 ---
    if (tid < T_) {
        float p   = pt[b * L_ + l0 + tid];
        float dcy = fmaxf(1.0f - p, EPS_);
        s_p[tid]  = p;
        s_S[tid]  = __logf(dcy);        // temporarily log(decay)
    }
    __syncthreads();
    if (tid == 0) {
        float acc = 0.0f;
        #pragma unroll
        for (int k = 0; k < T_; ++k) { acc += s_S[k]; s_S[k] = acc; s_P[k] = __expf(acc); }
    }
    __syncthreads();
    if (blockIdx.y == 0) {              // write once per (b,c)
        if (tid < T_) Pin[b * L_ + l0 + tid] = s_P[tid];
        if (tid == 0) Ptot[b * NC_ + c]      = s_P[T_ - 1];
    }

    // --- build A fragments: W rows 0..15 (top) and 16..31 (bottom), K=0..31 ---
    // ISA 16-bit A 16x32 layout: M = lane&15;
    //   lanes 0-15 : K(e) = e<8 ?  e    : e+8
    //   lanes 16-31: K(e) = e<8 ?  e+8  : e+16
    const int row_t = lo;
    const int row_b = lo + 16;
    const float S_t = s_S[row_t];
    const float S_b = s_S[row_b];
    v16h a_top, a_bot;
    #pragma unroll
    for (int e = 0; e < 16; ++e) {
        const int K  = (e < 8) ? (e + 8 * hi) : (e + 8 + 8 * hi);
        const float Sk = s_S[K];
        const float pk = s_p[K];
        const float wt = (row_t >= K) ? __expf(S_t - Sk) * pk : 0.0f;
        const float wb = (row_b >= K) ? __expf(S_b - Sk) * pk : 0.0f;
        a_top[e] = (_Float16)wt;
        a_bot[e] = (_Float16)wb;
    }

    // --- load B fragment: z_chunk (32 x 16 tile of D), column N = lane&15 ---
    // B layout (same K order as A): K(e) = e<8 ? e+8*hi : e+8+8*hi.
    const int    dcol = (blockIdx.y * 8 + wave) * 16 + lo;
    const float* zp   = z + ((size_t)(b * L_ + l0 + 8 * hi)) * D_ + dcol;
    v16h bz;
    #pragma unroll
    for (int e = 0; e < 16; ++e) bz[e] = (_Float16)zp[(size_t)(e < 8 ? e : e + 8) * D_];

    // --- two WMMAs: rows 0-15 and rows 16-31 of the chunk ---
    v8f c0 = {};
    v8f y_top = __builtin_amdgcn_wmma_f32_16x16x32_f16(false, a_top, false, bz,
                                                       (short)0, c0, false, false);
    v8f y_bot = __builtin_amdgcn_wmma_f32_16x16x32_f16(false, a_bot, false, bz,
                                                       (short)0, c0, false, false);

    // --- store: C/D layout: lanes 0-15 rows 0-7 (vgpr r -> M=r), lanes 16-31 rows 8-15 ---
    float* op = out + ((size_t)(b * L_ + l0)) * D_ + dcol;
    #pragma unroll
    for (int r = 0; r < 8; ++r) {
        op[(size_t)(8 * hi + r) * D_]        = y_top[r];
        op[(size_t)(16 + 8 * hi + r) * D_]   = y_bot[r];
    }
    if (hi) {  // row 31 lives in lanes 16-31, element 7 of y_bot
        yend[((size_t)b * NC_ + c) * D_ + dcol] = y_bot[7];
    }
}

// ---------------------------------------------------------------------------
// Kernel 2: sequential carry scan over chunks (f32).
//   Carry[b,c,d] = state before chunk c;  state' = Ptot*state + yend
// ---------------------------------------------------------------------------
__global__ void ema_scan_kernel(const float* __restrict__ Ptot,
                                const float* __restrict__ yend,
                                float* __restrict__ Carry)
{
    const int t = blockIdx.x * blockDim.x + threadIdx.x;
    if (t >= B_ * D_) return;
    const int b = t / D_;
    const int d = t - b * D_;
    float carry = 0.0f;
    for (int c = 0; c < NC_; ++c) {
        const size_t idx = ((size_t)b * NC_ + c) * D_ + d;
        Carry[idx] = carry;
        carry = Ptot[b * NC_ + c] * carry + yend[idx];
    }
}

// ---------------------------------------------------------------------------
// Kernel 3: apply inter-chunk carry: out += Carry[b,c(l),d] * Pin[b,l]
// ---------------------------------------------------------------------------
__global__ void ema_apply_kernel(const float* __restrict__ Carry,
                                 const float* __restrict__ Pin,
                                 float* __restrict__ out)
{
    const size_t i = (size_t)blockIdx.x * blockDim.x + threadIdx.x;
    const size_t total = (size_t)B_ * L_ * D_;
    if (i >= total) return;
    const size_t d  = i % D_;
    const size_t bl = i / D_;
    const size_t l  = bl % L_;
    const size_t b  = bl / L_;
    const size_t c  = l / T_;
    out[i] += Carry[((size_t)b * NC_ + c) * D_ + d] * Pin[b * L_ + l];
}

extern "C" void kernel_launch(void* const* d_in, const int* in_sizes, int n_in,
                              void* d_out, int out_size, void* d_ws, size_t ws_size,
                              hipStream_t stream)
{
    const float* z  = (const float*)d_in[0];   // [B, L, D] f32
    const float* pt = (const float*)d_in[1];   // [B, L]    f32
    float* out = (float*)d_out;                // [B, L, D] f32

    // Workspace partition (floats): Pin[B*L] | Ptot[B*NC] | yend[B*NC*D] | Carry[B*NC*D]
    float* Pin   = (float*)d_ws;
    float* Ptot  = Pin  + (size_t)B_ * L_;
    float* yend  = Ptot + (size_t)B_ * NC_;
    float* Carry = yend + (size_t)B_ * NC_ * D_;

    // Kernel 1: one block per (b, chunk) x 6 blocks over D (8 waves * 16 cols each)
    dim3 grid1(B_ * NC_, D_ / (16 * 8));
    ema_chunk_kernel<<<grid1, 256, 0, stream>>>(z, pt, out, Pin, Ptot, yend);

    // Kernel 2: B*D = 1536 independent carry scans
    ema_scan_kernel<<<(B_ * D_ + 255) / 256, 256, 0, stream>>>(Ptot, yend, Carry);

    // Kernel 3: elementwise carry application over B*L*D
    const size_t total = (size_t)B_ * L_ * D_;
    ema_apply_kernel<<<(int)((total + 255) / 256), 256, 0, stream>>>(Carry, Pin, out);
}
